// AD_Embedding_52046413693141
// MI455X (gfx1250) — matrix operand order repaired and checked by reading, not compiled, not run on hardware
//
#include <hip/hip_runtime.h>

typedef __attribute__((ext_vector_type(16))) _Float16 v16h;
typedef __attribute__((ext_vector_type(8)))  float    v8f;

#define BINS 30
#define EMB  100
#define WAVES 4
#define ROWS_PER_BLOCK 64   // 4 waves x 16 rows

union FragU { v16h v; uint4 q[2]; };

// A-fragment (16x32 f16): lane holds row M=lane&15, halves 0..7 at K=8*hi.., halves 8..15 at K=16+8*hi..
__device__ inline v16h lds_a_frag(const _Float16* rowbase, int hi) {
  FragU u;
  u.q[0] = *(const uint4*)(rowbase + 8*hi);
  u.q[1] = *(const uint4*)(rowbase + 16 + 8*hi);
  return u.v;
}
// B-fragment (32x16 f16): lane holds column N=lane&15, halves 0..7 at K=8*hi.., halves 8..15 at K=16+8*hi..
__device__ inline v16h lds_b_frag(const _Float16* colbase, int hi) {
  FragU u;
  u.q[0] = *(const uint4*)(colbase + 8*hi);
  u.q[1] = *(const uint4*)(colbase + 16 + 8*hi);
  return u.v;
}

__global__ __launch_bounds__(128) void fused_softmax_embed(
    const float* __restrict__ x,   // (B*F) scalars, one per row
    const float* __restrict__ W1,  // (30)
    const float* __restrict__ Wl,  // (30,30) row-major
    const float* __restrict__ W2,  // (100,30) row-major
    float* __restrict__ out)       // (B*F, 100) contiguous
{
  __shared__ __align__(16) _Float16 sWl[32*32];        // [n][k] = Wl[n][k], zero-padded
  __shared__ __align__(16) _Float16 sW2[112*32];       // [n][k] = W2[n][k], zero-padded
  __shared__ float sW1[32];
  __shared__ __align__(16) float    sZ[WAVES][16*32];  // matmul1 result per wave
  __shared__ __align__(16) _Float16 sA2[WAVES][16*32]; // exp() staging (A frag of matmul2)
  __shared__ float sRecip[WAVES][16];
  __shared__ float sX[WAVES][16];
  __shared__ __align__(16) float    sOut[WAVES][16*EMB];

  const int tid  = threadIdx.x;
  const int wave = tid >> 5;
  const int lane = tid & 31;
  const int lo   = lane & 15;
  const int hi   = lane >> 4;
  const int rowbase = blockIdx.x * ROWS_PER_BLOCK + wave * 16;

  // ---- stage weights into LDS (f16, zero-padded to 32-wide K) ----
  for (int i = tid; i < 32*32; i += 128) {
    int n = i >> 5, k = i & 31;
    sWl[i] = (n < BINS && k < BINS) ? (_Float16)Wl[n*BINS + k] : (_Float16)0.f;
  }
  for (int i = tid; i < 112*32; i += 128) {
    int n = i >> 5, k = i & 31;
    sW2[i] = (n < EMB && k < BINS) ? (_Float16)W2[n*BINS + k] : (_Float16)0.f;
  }
  if (tid < 32) sW1[tid] = (tid < BINS) ? W1[tid] : 0.f;
  if (lane < 16) sX[wave][lane] = x[rowbase + lane];
  __syncthreads();

  // ---- A1 fragment: y0[r][k] = leaky_relu(x_r * W1[k]) computed in registers ----
  const float xr = sX[wave][lo];
  v16h a1;
  #pragma unroll
  for (int i = 0; i < 8; i++) {
    float v = xr * sW1[8*hi + i];
    v = v > 0.f ? v : 0.01f * v;
    a1[i] = (_Float16)v;
    float v2 = xr * sW1[16 + 8*hi + i];
    v2 = v2 > 0.f ? v2 : 0.01f * v2;
    a1[8+i] = (_Float16)v2;
  }

  // ---- C fragments carry the residual 0.1*y0[row][n] for free ----
  v8f c0, c1;
  const float w1n0 = sW1[lo];
  const float w1n1 = sW1[lo + 16];
  #pragma unroll
  for (int j = 0; j < 8; j++) {
    float xj = sX[wave][j + 8*hi];
    float v0 = xj * w1n0; v0 = v0 > 0.f ? v0 : 0.01f * v0;
    float v1 = xj * w1n1; v1 = v1 > 0.f ? v1 : 0.01f * v1;
    c0[j] = 0.1f * v0;
    c1[j] = 0.1f * v1;
  }

  // ---- matmul1: z = y0 x Wl^T + 0.1*y0  (2 WMMAs, N tiles 0..15 / 16..31) ----
  v16h b0 = lds_b_frag(&sWl[lo*32], hi);
  v16h b1 = lds_b_frag(&sWl[(lo+16)*32], hi);
  v8f d0 = __builtin_amdgcn_wmma_f32_16x16x32_f16(false, a1, false, b0, (short)0, c0, false, false);
  v8f d1 = __builtin_amdgcn_wmma_f32_16x16x32_f16(false, a1, false, b1, (short)0, c1, false, false);
  #pragma unroll
  for (int j = 0; j < 8; j++) {
    sZ[wave][(j + 8*hi)*32 + lo]      = d0[j];
    sZ[wave][(j + 8*hi)*32 + lo + 16] = d1[j];
  }
  __syncthreads();

  // ---- softmax(0.5*z) per row; store unnormalized exp as f16, keep 1/sum ----
  if (lane < 16) {
    const float* zr = &sZ[wave][lane*32];
    float m = zr[0];
    for (int k = 1; k < BINS; k++) m = fmaxf(m, zr[k]);
    float sum = 0.f;
    _Float16* ar = &sA2[wave][lane*32];
    for (int k = 0; k < BINS; k++) {
      float e = __expf(0.5f * (zr[k] - m));
      sum += e;
      ar[k] = (_Float16)e;
    }
    ar[30] = (_Float16)0.f;   // K padding must be exact zero
    ar[31] = (_Float16)0.f;
    sRecip[wave][lane] = 1.f / sum;
  }
  __syncthreads();

  // ---- matmul2: out = (exp x W2^T) * (1/sum)  (7 WMMAs, N = 0..111, cols >= 100 dropped) ----
  v16h a2 = lds_a_frag(&sA2[wave][lo*32], hi);
  float rcp[8];
  #pragma unroll
  for (int j = 0; j < 8; j++) rcp[j] = sRecip[wave][j + 8*hi];
  #pragma unroll
  for (int t = 0; t < 7; t++) {
    v16h b2 = lds_b_frag(&sW2[(lo + 16*t)*32], hi);
    v8f cz = {0.f,0.f,0.f,0.f,0.f,0.f,0.f,0.f};
    v8f d = __builtin_amdgcn_wmma_f32_16x16x32_f16(false, a2, false, b2, (short)0, cz, false, false);
    int n = lo + 16*t;
    if (n < EMB) {
      #pragma unroll
      for (int j = 0; j < 8; j++)
        sOut[wave][(j + 8*hi)*EMB + n] = d[j] * rcp[j];
    }
  }
  __syncthreads();

  // ---- coalesced float4 store: tile is contiguous 1600 floats in output ----
  float4* dst = (float4*)(out + (size_t)rowbase * EMB);
  const float4* src = (const float4*)&sOut[wave][0];
  for (int i = lane; i < 400; i += 32) dst[i] = src[i];
}

extern "C" void kernel_launch(void* const* d_in, const int* in_sizes, int n_in,
                              void* d_out, int out_size, void* d_ws, size_t ws_size,
                              hipStream_t stream) {
  const float* x  = (const float*)d_in[0];   // (16384,1,30)
  const float* W1 = (const float*)d_in[1];   // (30,1)
  const float* Wl = (const float*)d_in[2];   // (30,30)
  const float* W2 = (const float*)d_in[3];   // (100,30)
  float* out = (float*)d_out;                // (16384, 3000)

  const int total_rows = 16384 * 30;                 // 491520
  dim3 grid(total_rows / ROWS_PER_BLOCK);            // 7680 blocks
  fused_softmax_embed<<<grid, 128, 0, stream>>>(x, W1, Wl, W2, out);
}
